// Graphormer_83683142795593
// MI455X (gfx1250) — matrix operand order repaired and checked by reading, not compiled, not run on hardware
//
#include <hip/hip_runtime.h>
#include <hip/hip_bf16.h>
#include <stdint.h>

// ---------------------------------------------------------------------------
// Graphormer block (2x GAT + LN + FFN + LN) for MI455X (gfx1250, wave32).
// bf16 WMMA (v_wmma_f32_16x16x32_bf16) for every GEMM; 128x128 block tiles,
// double-buffered LDS. A tile staged with GLOBAL_LOAD_ASYNC_TO_LDS_B128
// (ASYNCcnt) -- no VGPR round trip; B tile register-transposed so both A and
// B fragments are gathered with ds_load_b128. Column softmax fused with adj
// masking + leaky-relu; softmax normalization folded into Wh row scale.
// ---------------------------------------------------------------------------

typedef __attribute__((ext_vector_type(16))) __bf16 v16bf;
typedef __attribute__((ext_vector_type(8)))  __bf16 v8bf;
typedef __attribute__((ext_vector_type(2)))  __bf16 v2bf;
typedef __attribute__((ext_vector_type(4)))  __bf16 v4bf;
typedef __attribute__((ext_vector_type(8)))  float  v8f;

#define GAT_ALPHA 0.2f
#define GAT_NEG  -9.0e15f
#define LN_EPS    1e-5f

#define BM 128
#define BN 128
#define BK 32
#define LDA (BK + 8)    // 40 elems = 80B row stride (16B multiple)
#define LDB (BK + 8)    // transposed B: row = output column

// async global->LDS b128: vdst = per-lane LDS byte offset, vaddr = 64b global
__device__ __forceinline__ void async_load_b128(unsigned lds_off,
                                                const void* gaddr) {
    asm volatile("global_load_async_to_lds_b128 %0, %1, off"
                 :: "v"(lds_off), "v"((unsigned long long)(uintptr_t)gaddr)
                 : "memory");
}
__device__ __forceinline__ void wait_async0() {
    asm volatile("s_wait_asynccnt 0x0" ::: "memory");
}

// ---------------------------------------------------------------------------
// f32 -> bf16 conversion (vector-of-4 per thread)
// ---------------------------------------------------------------------------
__global__ __launch_bounds__(256)
void cvt_f32_to_bf16(const float* __restrict__ in, __bf16* __restrict__ out,
                     long long n) {
    long long i = ((long long)blockIdx.x * blockDim.x + threadIdx.x) * 4;
    if (i + 3 < n) {
        float4 v = *(const float4*)(in + i);
        v4bf o;
        o[0] = (__bf16)v.x; o[1] = (__bf16)v.y;
        o[2] = (__bf16)v.z; o[3] = (__bf16)v.w;
        *(v4bf*)(out + i) = o;
    }
}

// ---------------------------------------------------------------------------
// Generic bf16 WMMA GEMM:  C[b] = A[b] @ B[b]  (row-major, dims divide tiles)
//  - 256 threads = 8 waves; block tile 128x128, K-step 32, LDS double-buffer
//  - wave grid 4(M) x 2(N); each wave owns 32x64 = 2x4 WMMA tiles
//  - A staged via async global->LDS b128 (ASYNCcnt); B staged transposed
//  - optional bias (per output column), optional ReLU
//  - writes f32 and/or bf16 outputs (either pointer may be null)
// ---------------------------------------------------------------------------
__global__ __launch_bounds__(256)
void gemm_bf16_wmma(const __bf16* __restrict__ A, const __bf16* __restrict__ B,
                    float* __restrict__ Cf, __bf16* __restrict__ Cb,
                    const float* __restrict__ bias, int relu,
                    int N, int K,
                    long long strideA, long long strideB, long long strideC) {
    __shared__ __align__(16) __bf16 As[2][BM][LDA];   // row-major A tile
    __shared__ __align__(16) __bf16 Bt[2][BN][LDB];   // TRANSPOSED B: [col][k]

    const int tid   = threadIdx.x;
    const int lane  = tid & 31;
    const int wid   = tid >> 5;       // 0..7
    const int waveM = wid & 3;        // 4 waves along M (32 rows each)
    const int waveN = wid >> 2;       // 2 waves along N (64 cols each)

    const int bM    = blockIdx.x * BM;
    const int bN    = blockIdx.y * BN;
    const int batch = blockIdx.z;

    const __bf16* Ab = A + (long long)batch * strideA;
    const __bf16* Bb = B + (long long)batch * strideB;

    v8f acc[2][4] = {};

    // --- staging indices ---
    // A tile 128x32: 2x async b128 per thread
    const int arow = tid >> 2;            // 0..63 (+64 second half)
    const int acol = (tid & 3) * 8;       // 0,8,16,24
    // B tile 32x128: row-pairs; 2x b128 per thread, stored transposed
    const int brp  = tid >> 4;            // 0..15 -> rows 2*brp, 2*brp+1
    const int bc8  = (tid & 15) * 8;      // col group of 8

    // --- fragment gather indices (wave32; contiguous k = 16h + s for A and B) ---
    const int am  = lane & 15;
    const int akb = (lane < 16) ? 0 : 16;  // A: contiguous k = 16h + s (matches B)
    const int bkb = (lane < 16) ? 0 : 16;  // B: contiguous k = 16h + s (matches A)
    const int bn  = lane & 15;

    const int nk = K / BK;

    // ---- prologue: stage tile 0 into buffer 0 ----
    {
        async_load_b128((unsigned)(uintptr_t)&As[0][arow][acol],
                        Ab + (long long)(bM + arow) * K + acol);
        async_load_b128((unsigned)(uintptr_t)&As[0][arow + 64][acol],
                        Ab + (long long)(bM + arow + 64) * K + acol);
        v8bf rb0 = *(const v8bf*)(Bb + (long long)(2 * brp)     * N + bN + bc8);
        v8bf rb1 = *(const v8bf*)(Bb + (long long)(2 * brp + 1) * N + bN + bc8);
        #pragma unroll
        for (int e = 0; e < 8; ++e) {
            v2bf p; p[0] = rb0[e]; p[1] = rb1[e];
            *(v2bf*)(&Bt[0][bc8 + e][2 * brp]) = p;
        }
    }

    int cur = 0;
    for (int kt = 0; kt < nk; ++kt) {
        wait_async0();     // my async A loads have landed in LDS
        __syncthreads();   // everyone's tile[cur] ready; tile[cur^1] free

        // ---- stage tile kt+1 into the other buffer (overlaps WMMAs) ----
        if (kt + 1 < nk) {
            const int k0  = (kt + 1) * BK;
            const int nxt = cur ^ 1;
            async_load_b128((unsigned)(uintptr_t)&As[nxt][arow][acol],
                            Ab + (long long)(bM + arow) * K + k0 + acol);
            async_load_b128((unsigned)(uintptr_t)&As[nxt][arow + 64][acol],
                            Ab + (long long)(bM + arow + 64) * K + k0 + acol);
            v8bf rb0 = *(const v8bf*)(Bb + (long long)(k0 + 2 * brp)     * N + bN + bc8);
            v8bf rb1 = *(const v8bf*)(Bb + (long long)(k0 + 2 * brp + 1) * N + bN + bc8);
            #pragma unroll
            for (int e = 0; e < 8; ++e) {
                v2bf p; p[0] = rb0[e]; p[1] = rb1[e];
                *(v2bf*)(&Bt[nxt][bc8 + e][2 * brp]) = p;
            }
            if (kt + 2 < nk) {
                __builtin_prefetch(Bb + (long long)(k0 + BK + 2 * brp) * N + bN + bc8, 0, 1);
            }
        }

        // ---- gather fragments from buf[cur] (all 2x ds_load_b128) ----
        v16bf afrag[2], bfrag[4];
        #pragma unroll
        for (int t = 0; t < 2; ++t) {
            const int r = waveM * 32 + t * 16 + am;
            #pragma unroll
            for (int j = 0; j < 16; ++j) {
                const int kk = akb + j;
                afrag[t][j] = As[cur][r][kk];
            }
        }
        #pragma unroll
        for (int t = 0; t < 4; ++t) {
            const int c = waveN * 64 + t * 16 + bn;
            #pragma unroll
            for (int j = 0; j < 16; ++j)
                bfrag[t][j] = Bt[cur][c][bkb + j];
        }

        // ---- 8 WMMAs per k-step ----
        #pragma unroll
        for (int i = 0; i < 2; ++i)
            #pragma unroll
            for (int j = 0; j < 4; ++j)
                acc[i][j] = __builtin_amdgcn_wmma_f32_16x16x32_bf16(
                    false, afrag[i], false, bfrag[j],
                    (short)0, acc[i][j], false, false);

        cur ^= 1;
    }

    // ---- epilogue: bias / relu / f32 and-or bf16 stores ----
    #pragma unroll
    for (int i = 0; i < 2; ++i) {
        #pragma unroll
        for (int j = 0; j < 4; ++j) {
            const int col  = bN + waveN * 64 + j * 16 + (lane & 15);
            const float bv = bias ? bias[col] : 0.0f;
            #pragma unroll
            for (int r = 0; r < 8; ++r) {
                const int rowl = r + ((lane < 16) ? 0 : 8);
                const int grow = bM + waveM * 32 + i * 16 + rowl;
                float v = acc[i][j][r] + bv;
                if (relu) v = v > 0.0f ? v : 0.0f;
                const long long idx = (long long)batch * strideC
                                    + (long long)grow * N + col;
                if (Cf) Cf[idx] = v;
                if (Cb) Cb[idx] = (__bf16)v;
            }
        }
    }
}

// ---------------------------------------------------------------------------
// Attention logits: Wh1[r] = Wh[r,:].a[0:F], Wh2[r] = Wh[r,:].a[F:2F]
// ---------------------------------------------------------------------------
__global__ __launch_bounds__(256)
void attn_logits(const float* __restrict__ Wh, const float* __restrict__ a,
                 float* __restrict__ Wh1, float* __restrict__ Wh2, int F) {
    __shared__ float s1[256];
    __shared__ float s2[256];
    const long long row = blockIdx.x;
    const int t = threadIdx.x;
    const float v = Wh[row * F + t];
    s1[t] = v * a[t];
    s2[t] = v * a[F + t];
    __syncthreads();
    for (int off = 128; off > 0; off >>= 1) {
        if (t < off) { s1[t] += s1[t + off]; s2[t] += s2[t + off]; }
        __syncthreads();
    }
    if (t == 0) { Wh1[row] = s1[0]; Wh2[row] = s2[0]; }
}

// ---------------------------------------------------------------------------
// Column softmax over scores[i,j] = mask(adj, leaky_relu(Wh1[i]+Wh2[j])).
// Thread <-> column j: coalesced adj/att. Writes unnormalized exp() in bf16
// att and 1/colsum (normalization folded into Wh row-scaling -- exact).
// ---------------------------------------------------------------------------
__global__ __launch_bounds__(256)
void gat_softmax_col(const float* __restrict__ Wh1, const float* __restrict__ Wh2,
                     const int* __restrict__ adj, __bf16* __restrict__ att,
                     float* __restrict__ colinv, int N) {
    const int b = blockIdx.z;
    const int j = blockIdx.x * 256 + threadIdx.x;
    const int*   adjb = adj + (long long)b * N * N;
    const float* w1   = Wh1 + (long long)b * N;
    __bf16*      attb = att + (long long)b * N * N;
    const float  w2   = Wh2[(long long)b * N + j];

    float m = -3.0e38f;
    #pragma unroll 4
    for (int i = 0; i < N; ++i) {
        float e = w1[i] + w2;
        e = e > 0.0f ? e : GAT_ALPHA * e;
        const float s = (adjb[(long long)i * N + j] > 0) ? e : GAT_NEG;
        m = fmaxf(m, s);
    }
    float sum = 0.0f;
    #pragma unroll 4
    for (int i = 0; i < N; ++i) {
        float e = w1[i] + w2;
        e = e > 0.0f ? e : GAT_ALPHA * e;
        const float s = (adjb[(long long)i * N + j] > 0) ? e : GAT_NEG;
        const float p = __expf(s - m);
        sum += p;
        attb[(long long)i * N + j] = (__bf16)p;
    }
    colinv[(long long)b * N + j] = 1.0f / sum;
}

// ---------------------------------------------------------------------------
// Whs[r,:] = bf16(Wh[r,:] * colinv[r])   (folds softmax denominator into Wh)
// ---------------------------------------------------------------------------
__global__ __launch_bounds__(256)
void scale_rows_to_bf16(const float* __restrict__ Wh, const float* __restrict__ colinv,
                        __bf16* __restrict__ out, int F) {
    const long long row = blockIdx.x;
    const int t = threadIdx.x;
    const float s = colinv[row];
    out[row * F + t] = (__bf16)(Wh[row * F + t] * s);
}

// ---------------------------------------------------------------------------
// y = LayerNorm(res + h) * g + b ; one block per row (F == 256)
// ---------------------------------------------------------------------------
__global__ __launch_bounds__(256)
void layernorm_residual(const float* __restrict__ res, const float* __restrict__ h,
                        const float* __restrict__ g, const float* __restrict__ bb,
                        float* __restrict__ outf, __bf16* __restrict__ outb, int F) {
    __shared__ float red[256];
    const long long row = blockIdx.x;
    const int t = threadIdx.x;
    const float v = res[row * F + t] + h[row * F + t];

    red[t] = v;
    __syncthreads();
    for (int off = 128; off > 0; off >>= 1) {
        if (t < off) red[t] += red[t + off];
        __syncthreads();
    }
    const float mu = red[0] / (float)F;
    __syncthreads();

    const float d = v - mu;
    red[t] = d * d;
    __syncthreads();
    for (int off = 128; off > 0; off >>= 1) {
        if (t < off) red[t] += red[t + off];
        __syncthreads();
    }
    const float var = red[0] / (float)F;
    __syncthreads();

    const float y = d * rsqrtf(var + LN_EPS) * g[t] + bb[t];
    if (outf) outf[row * F + t] = y;
    if (outb) outb[row * F + t] = (__bf16)y;
}

// ---------------------------------------------------------------------------
// Host-side orchestration
// ---------------------------------------------------------------------------
extern "C" void kernel_launch(void* const* d_in, const int* in_sizes, int n_in,
                              void* d_out, int out_size, void* d_ws, size_t ws_size,
                              hipStream_t stream) {
    (void)in_sizes; (void)n_in; (void)out_size; (void)ws_size;

    const int B = 2, N = 4096, F = 256, H = 1024;
    const int M = B * N;                       // 8192 rows

    const float* x     = (const float*)d_in[0];
    const int*   adj   = (const int*)  d_in[1];
    const float* W1    = (const float*)d_in[2];
    const float* a1    = (const float*)d_in[3];
    const float* W2    = (const float*)d_in[4];
    const float* a2    = (const float*)d_in[5];
    const float* wff1  = (const float*)d_in[6];
    const float* bff1  = (const float*)d_in[7];
    const float* wff2  = (const float*)d_in[8];
    const float* bff2  = (const float*)d_in[9];
    const float* gln1  = (const float*)d_in[10];
    const float* bln1  = (const float*)d_in[11];
    const float* gln2  = (const float*)d_in[12];
    const float* bln2  = (const float*)d_in[13];
    float* out = (float*)d_out;

    // workspace carve-up (256B aligned)
    char* w = (char*)d_ws;
    size_t off = 0;
    auto alloc = [&](size_t bytes) -> void* {
        void* p = w + off;
        off = (off + bytes + 255) & ~(size_t)255;
        return p;
    };
    __bf16* x_bf   = (__bf16*)alloc((size_t)M * F * 2);
    __bf16* W1b    = (__bf16*)alloc((size_t)F * F * 2);
    __bf16* W2b    = (__bf16*)alloc((size_t)F * F * 2);
    __bf16* Wf1b   = (__bf16*)alloc((size_t)F * H * 2);
    __bf16* Wf2b   = (__bf16*)alloc((size_t)H * F * 2);
    float*  Wh_f   = (float*) alloc((size_t)M * F * 4);
    float*  Wh1    = (float*) alloc((size_t)M * 4);
    float*  Wh2    = (float*) alloc((size_t)M * 4);
    float*  colinv = (float*) alloc((size_t)M * 4);
    __bf16* att    = (__bf16*)alloc((size_t)B * N * N * 2);
    __bf16* Whs    = (__bf16*)alloc((size_t)M * F * 2);
    float*  h1_f   = (float*) alloc((size_t)M * F * 4);
    __bf16* h1_bf  = (__bf16*)alloc((size_t)M * F * 2);
    float*  h2_f   = (float*) alloc((size_t)M * F * 4);
    float*  y_f    = (float*) alloc((size_t)M * F * 4);
    __bf16* y_bf   = (__bf16*)alloc((size_t)M * F * 2);
    __bf16* t_bf   = (__bf16*)alloc((size_t)M * H * 2);
    float*  ff_f   = (float*) alloc((size_t)M * F * 4);

    const dim3 blk(256);
    auto cvt = [&](const float* in, __bf16* o, long long n) {
        cvt_f32_to_bf16<<<dim3((unsigned)((n / 4 + 255) / 256)), blk, 0, stream>>>(in, o, n);
    };
    auto gemm = [&](const __bf16* Ap, const __bf16* Bp, float* Cf, __bf16* Cb,
                    const float* bias, int relu, int Mm, int Nn, int Kk,
                    int batches, long long sA, long long sB, long long sC) {
        dim3 grid((unsigned)(Mm / BM), (unsigned)(Nn / BN), (unsigned)batches);
        gemm_bf16_wmma<<<grid, blk, 0, stream>>>(Ap, Bp, Cf, Cb, bias, relu,
                                                 Nn, Kk, sA, sB, sC);
    };

    // ---- precision staging ----
    cvt(x,    x_bf,  (long long)M * F);
    cvt(W1,   W1b,   (long long)F * F);
    cvt(W2,   W2b,   (long long)F * F);
    cvt(wff1, Wf1b,  (long long)F * H);
    cvt(wff2, Wf2b,  (long long)H * F);

    // ---- GAT layer 1 ----
    gemm(x_bf, W1b, Wh_f, nullptr, nullptr, 0, M, F, F, 1, 0, 0, 0);
    attn_logits<<<dim3(M), blk, 0, stream>>>(Wh_f, a1, Wh1, Wh2, F);
    gat_softmax_col<<<dim3(N / 256, 1, B), blk, 0, stream>>>(Wh1, Wh2, adj, att, colinv, N);
    scale_rows_to_bf16<<<dim3(M), blk, 0, stream>>>(Wh_f, colinv, Whs, F);
    gemm(att, Whs, h1_f, h1_bf, nullptr, 0, N, F, N, B,
         (long long)N * N, (long long)N * F, (long long)N * F);

    // ---- GAT layer 2 ----
    gemm(h1_bf, W2b, Wh_f, nullptr, nullptr, 0, M, F, F, 1, 0, 0, 0);
    attn_logits<<<dim3(M), blk, 0, stream>>>(Wh_f, a2, Wh1, Wh2, F);
    gat_softmax_col<<<dim3(N / 256, 1, B), blk, 0, stream>>>(Wh1, Wh2, adj, att, colinv, N);
    scale_rows_to_bf16<<<dim3(M), blk, 0, stream>>>(Wh_f, colinv, Whs, F);
    gemm(att, Whs, h2_f, nullptr, nullptr, 0, N, F, N, B,
         (long long)N * N, (long long)N * F, (long long)N * F);

    // ---- LN1(x + h2) ----
    layernorm_residual<<<dim3(M), blk, 0, stream>>>(x, h2_f, gln1, bln1, y_f, y_bf, F);

    // ---- FFN ----
    gemm(y_bf, Wf1b, nullptr, t_bf, bff1, 1, M, H, F, 1, 0, 0, 0);   // relu(y@W1+b1) -> bf16
    gemm(t_bf, Wf2b, ff_f, nullptr, bff2, 0, M, F, H, 1, 0, 0, 0);   // @W2+b2 -> f32

    // ---- LN2(y + ff) -> out ----
    layernorm_residual<<<dim3(M), blk, 0, stream>>>(y_f, ff_f, gln2, bln2, out, nullptr, F);
}
